// ResidualNegativeMamba_17437567222401
// MI455X (gfx1250) — compile-verified, matrix operand order checked by reading, not run on hardware
//
#include <hip/hip_runtime.h>
#include <hip/hip_fp16.h>

typedef __attribute__((ext_vector_type(16))) _Float16 v16h;
typedef __attribute__((ext_vector_type(8)))  float    v8f;

#define BB    2
#define LL    2048
#define DM    768
#define DI    1536
#define DS    16
#define DCONV 4
#define DTR   48
#define MTOT  (BB * LL)          // 4096 rows
#define NXP   128                // x_proj N padded 80 -> 128 (N%128==0 for all GEMMs)
#define KDT   64                 // dt_proj K padded 48 -> 64
#define NCH   32                 // scan chunks
#define CL    (LL / NCH)         // 64 steps per chunk

// ---------------------------------------------------------------------------
// WMMA fragment helpers (CDNA5 16x16x32 f16, wave32; ISA 7.12.2 layouts)
// A (16x32, MxK): lane holds row m = lane&15. lanes 0-15: K in {0..7,16..23};
// lanes 16-31: K in {8..15,24..31}. VGPR v holds K-pair:
//   k = ((v&4)?16:0) + 2*(v&3) + ((lane&16)?8:0)
// B fragment from weights stored (N,K) row-major uses the identical mapping
// with n = lane&15 (contiguous in K, same as A).
// ---------------------------------------------------------------------------
__device__ __forceinline__ v16h load_frag(const _Float16* __restrict__ base,
                                          int row, int ld, int kTile, int lane) {
  const _Float16* p = base + (size_t)row * ld + kTile + ((lane & 16) ? 8 : 0);
  v16h f;
#pragma unroll
  for (int v = 0; v < 8; ++v) {
    int k = ((v & 4) ? 16 : 0) + 2 * (v & 3);
    f[2 * v]     = p[k];
    f[2 * v + 1] = p[k + 1];
  }
  return f;
}

// C/D (16x16 f32, 8 VGPRs): VGPR r -> m = r + ((lane&16)?8:0), n = lane&15
// mode 0: plain; mode 1: softplus(acc + bias[n]); mode 2: acc + resid[m,n]
__device__ __forceinline__ void store_tile(v8f acc, float* __restrict__ C,
                                           int m0, int n0, int ldc, int lane,
                                           int mode, const float* __restrict__ aux) {
  int n = n0 + (lane & 15);
  int mh = (lane & 16) ? 8 : 0;
#pragma unroll
  for (int r = 0; r < 8; ++r) {
    int m = m0 + mh + r;
    float v = acc[r];
    if (mode == 1) {
      v += aux[n];
      v = (v > 20.f) ? v : log1pf(__expf(v));          // softplus
    } else if (mode == 2) {
      v += aux[(size_t)m * ldc + n];                    // residual
    }
    C[(size_t)m * ldc + n] = v;
  }
}

// ---------------------------------------------------------------------------
// Tiled WMMA GEMM: C[M,N] = A[M,K] * W[N,K]^T (+ epilogue)
// 128 threads = 4 waves arranged 2x2; wave tile 32x64 -> block tile 64x128.
// Per K-step each wave: 2 A frags + 4 B frags -> 8 v_wmma (A reused 4x, B 2x).
// Requires M%64==0, N%128==0, K%32==0 (guaranteed by padding).
// ---------------------------------------------------------------------------
__global__ void __launch_bounds__(128)
wmma_gemm(const _Float16* __restrict__ A, const _Float16* __restrict__ W,
          float* __restrict__ C, int M, int N, int K,
          int mode, const float* __restrict__ aux) {
  const int lane = threadIdx.x & 31;
  const int wave = threadIdx.x >> 5;
  const int mOff = blockIdx.y * 64 + (wave & 1) * 32;
  const int nOff = blockIdx.x * 128 + (wave >> 1) * 64;
  if (mOff >= M || nOff >= N) return;   // wave-uniform (EXEC stays full)

  v8f acc[2][4];
#pragma unroll
  for (int i = 0; i < 2; ++i)
#pragma unroll
    for (int j = 0; j < 4; ++j) acc[i][j] = (v8f){};

  const int aRow = mOff + (lane & 15);
  const int bRow = nOff + (lane & 15);

  for (int k = 0; k < K; k += 32) {
    // prefetch next K tile into cache (global_prefetch_b8)
    if (k + 64 < K) {
      __builtin_prefetch(A + (size_t)aRow * K + k + 64, 0, 1);
      __builtin_prefetch(W + (size_t)bRow * K + k + 64, 0, 1);
    }
    v16h a0 = load_frag(A, aRow,      K, k, lane);
    v16h a1 = load_frag(A, aRow + 16, K, k, lane);
    v16h b0 = load_frag(W, bRow,      K, k, lane);
    v16h b1 = load_frag(W, bRow + 16, K, k, lane);
    v16h b2 = load_frag(W, bRow + 32, K, k, lane);
    v16h b3 = load_frag(W, bRow + 48, K, k, lane);
    acc[0][0] = __builtin_amdgcn_wmma_f32_16x16x32_f16(false, a0, false, b0, (short)0, acc[0][0], false, false);
    acc[0][1] = __builtin_amdgcn_wmma_f32_16x16x32_f16(false, a0, false, b1, (short)0, acc[0][1], false, false);
    acc[0][2] = __builtin_amdgcn_wmma_f32_16x16x32_f16(false, a0, false, b2, (short)0, acc[0][2], false, false);
    acc[0][3] = __builtin_amdgcn_wmma_f32_16x16x32_f16(false, a0, false, b3, (short)0, acc[0][3], false, false);
    acc[1][0] = __builtin_amdgcn_wmma_f32_16x16x32_f16(false, a1, false, b0, (short)0, acc[1][0], false, false);
    acc[1][1] = __builtin_amdgcn_wmma_f32_16x16x32_f16(false, a1, false, b1, (short)0, acc[1][1], false, false);
    acc[1][2] = __builtin_amdgcn_wmma_f32_16x16x32_f16(false, a1, false, b2, (short)0, acc[1][2], false, false);
    acc[1][3] = __builtin_amdgcn_wmma_f32_16x16x32_f16(false, a1, false, b3, (short)0, acc[1][3], false, false);
  }
#pragma unroll
  for (int i = 0; i < 2; ++i)
#pragma unroll
    for (int j = 0; j < 4; ++j)
      store_tile(acc[i][j], C, mOff + i * 16, nOff + j * 16, N, lane, mode, aux);
}

// ---------------------------------------------------------------------------
// Weight convert f32 -> f16 with zero padding of rows/cols
// ---------------------------------------------------------------------------
__global__ void convert_pad(const float* __restrict__ src, _Float16* __restrict__ dst,
                            int srows, int scols, int drows, int dcols) {
  long long i = (long long)blockIdx.x * blockDim.x + threadIdx.x;
  long long tot = (long long)drows * dcols;
  if (i >= tot) return;
  int c = (int)(i % dcols);
  long long r = i / dcols;
  float v = (r < srows && c < scols) ? src[r * scols + c] : 0.f;
  dst[i] = (_Float16)v;
}

// A_neg[d*16+n] = -exp(A_log[d*16+n])  (hoisted out of scan passes)
__global__ void neg_exp_kernel(const float* __restrict__ A_log, float* __restrict__ An) {
  int i = blockIdx.x * blockDim.x + threadIdx.x;
  if (i < DI * DS) An[i] = -__expf(A_log[i]);
}

// ---------------------------------------------------------------------------
// RMSNorm: xn = x * rsqrt(mean(x^2)+eps) * w, f16 output (GEMM A operand)
// ---------------------------------------------------------------------------
__global__ void __launch_bounds__(256)
rmsnorm_kernel(const float* __restrict__ x, const float* __restrict__ w,
               _Float16* __restrict__ xn) {
  const int row = blockIdx.x;
  const float* xr = x + (size_t)row * DM;
  float s = 0.f;
  for (int i = threadIdx.x; i < DM; i += 256) { float v = xr[i]; s += v * v; }
  __shared__ float red[256];
  red[threadIdx.x] = s;
  __syncthreads();
  for (int off = 128; off > 0; off >>= 1) {
    if (threadIdx.x < off) red[threadIdx.x] += red[threadIdx.x + off];
    __syncthreads();
  }
  float scale = rsqrtf(red[0] / (float)DM + 1e-5f);
  for (int i = threadIdx.x; i < DM; i += 256)
    xn[(size_t)row * DM + i] = (_Float16)(xr[i] * scale * w[i]);
}

// ---------------------------------------------------------------------------
// Causal depthwise conv (k=4, left pad 3, per-batch) + bias + SiLU -> f16
// ---------------------------------------------------------------------------
__global__ void __launch_bounds__(256)
conv_silu_kernel(const float* __restrict__ xz, const float* __restrict__ cw,
                 const float* __restrict__ cb, _Float16* __restrict__ xs_h) {
  long long i = (long long)blockIdx.x * blockDim.x + threadIdx.x;
  if (i >= (long long)MTOT * DI) return;
  int d = (int)(i % DI);
  long long row = i / DI;
  int l = (int)(row % LL);
  long long b0 = row - l;                      // batch base row
  float s = cb[d];
#pragma unroll
  for (int k = 0; k < DCONV; ++k) {
    int ll = l - (DCONV - 1) + k;
    if (ll >= 0) s += cw[d * DCONV + k] * xz[(b0 + ll) * (2 * DI) + d];
  }
  float sil = s / (1.f + __expf(-s));
  xs_h[i] = (_Float16)sil;
}

// ---------------------------------------------------------------------------
// Repack delta_raw = dBC[:, 0:48] into f16 M x 64 (zero-padded K for dt_proj)
// ---------------------------------------------------------------------------
__global__ void __launch_bounds__(256)
pack_delta_kernel(const float* __restrict__ dBC, _Float16* __restrict__ dh) {
  long long i = (long long)blockIdx.x * blockDim.x + threadIdx.x;
  if (i >= (long long)MTOT * KDT) return;
  int c = (int)(i % KDT);
  long long r = i / KDT;
  float v = (c < DTR) ? dBC[r * NXP + c] : 0.f;
  dh[i] = (_Float16)v;
}

// ---------------------------------------------------------------------------
// Chunked selective scan (linear recurrence h_t = a_t h_{t-1} + b_t):
//  pass1: per (b,d,chunk) compute chunk summaries aprod[16], hsum[16]
//  pass2: per (b,d) 32-step stitch -> per-chunk initial states
//  pass3: per (b,d,chunk) local rescan with init; y = <h,C> + D*xs; SiLU gate
// 32x more parallelism than a monolithic scan (98304 vs 3072 threads).
// ---------------------------------------------------------------------------
__global__ void __launch_bounds__(256)
scan_pass1(const float* __restrict__ delta, const _Float16* __restrict__ xs_h,
           const float* __restrict__ dBC, const float* __restrict__ An,
           float* __restrict__ Aprod, float* __restrict__ Hsum) {
  int t = blockIdx.x * blockDim.x + threadIdx.x;
  if (t >= BB * NCH * DI) return;
  int d = t % DI;
  int bc = t / DI;
  int c = bc % NCH, b = bc / NCH;
  float a[DS], h[DS], ap[DS];
#pragma unroll
  for (int n = 0; n < DS; ++n) { a[n] = An[d * DS + n]; h[n] = 0.f; ap[n] = 1.f; }
  for (int i = 0; i < CL; ++i) {
    size_t row = (size_t)b * LL + c * CL + i;
    float dv  = delta[row * DI + d];
    float xs  = (float)xs_h[row * DI + d];
    float dxs = dv * xs;
    const float* bcp = dBC + row * NXP;
#pragma unroll
    for (int n = 0; n < DS; ++n) {
      float dA = __expf(dv * a[n]) * 2.f - 1.f;
      ap[n] *= dA;
      h[n] = dA * h[n] + dxs * bcp[DTR + n];
    }
  }
  size_t o = ((size_t)(b * DI + d) * NCH + c) * DS;
#pragma unroll
  for (int n = 0; n < DS; ++n) { Aprod[o + n] = ap[n]; Hsum[o + n] = h[n]; }
}

__global__ void __launch_bounds__(256)
scan_pass2(const float* __restrict__ Aprod, const float* __restrict__ Hsum,
           float* __restrict__ Hinit) {
  int t = blockIdx.x * blockDim.x + threadIdx.x;
  if (t >= BB * DI) return;
  float hin[DS];
#pragma unroll
  for (int n = 0; n < DS; ++n) hin[n] = 0.f;
  size_t base = (size_t)t * NCH * DS;
  for (int c = 0; c < NCH; ++c) {
    size_t o = base + (size_t)c * DS;
#pragma unroll
    for (int n = 0; n < DS; ++n) {
      Hinit[o + n] = hin[n];
      hin[n] = Aprod[o + n] * hin[n] + Hsum[o + n];
    }
  }
}

__global__ void __launch_bounds__(256)
scan_pass3(const float* __restrict__ delta, const _Float16* __restrict__ xs_h,
           const float* __restrict__ xz, const float* __restrict__ dBC,
           const float* __restrict__ An, const float* __restrict__ Dv,
           const float* __restrict__ Hinit, _Float16* __restrict__ y_h) {
  int t = blockIdx.x * blockDim.x + threadIdx.x;
  if (t >= BB * NCH * DI) return;
  int d = t % DI;
  int bc = t / DI;
  int c = bc % NCH, b = bc / NCH;
  float a[DS], h[DS];
  size_t o = ((size_t)(b * DI + d) * NCH + c) * DS;
#pragma unroll
  for (int n = 0; n < DS; ++n) { a[n] = An[d * DS + n]; h[n] = Hinit[o + n]; }
  float Dd = Dv[d];
  for (int i = 0; i < CL; ++i) {
    size_t row = (size_t)b * LL + c * CL + i;
    float dv  = delta[row * DI + d];
    float xs  = (float)xs_h[row * DI + d];
    float zv  = xz[row * (2 * DI) + DI + d];
    float dxs = dv * xs;
    const float* bcp = dBC + row * NXP;
    float y = 0.f;
#pragma unroll
    for (int n = 0; n < DS; ++n) {
      float dA = __expf(dv * a[n]) * 2.f - 1.f;
      h[n] = dA * h[n] + dxs * bcp[DTR + n];
      y += h[n] * bcp[DTR + DS + n];
    }
    y += Dd * xs;
    float sz = zv / (1.f + __expf(-zv));
    y_h[row * DI + d] = (_Float16)(y * sz);
  }
}

// ---------------------------------------------------------------------------
extern "C" void kernel_launch(void* const* d_in, const int* in_sizes, int n_in,
                              void* d_out, int out_size, void* d_ws, size_t ws_size,
                              hipStream_t stream) {
  const float* x        = (const float*)d_in[0];
  const float* in_w     = (const float*)d_in[1];
  const float* conv_w   = (const float*)d_in[2];
  const float* conv_b   = (const float*)d_in[3];
  const float* xproj_w  = (const float*)d_in[4];
  const float* dt_w     = (const float*)d_in[5];
  const float* dt_b     = (const float*)d_in[6];
  const float* A_log    = (const float*)d_in[7];
  const float* Dvec     = (const float*)d_in[8];
  const float* out_w    = (const float*)d_in[9];
  const float* rms_w    = (const float*)d_in[10];
  float* out = (float*)d_out;

  char* ws = (char*)d_ws;
  size_t off = 0;
  auto alloc = [&](size_t bytes) -> void* {
    void* p = ws + off;
    off = (off + bytes + 255) & ~(size_t)255;
    return p;
  };
  _Float16* xn_h    = (_Float16*)alloc((size_t)MTOT * DM * 2);
  _Float16* w_in_h  = (_Float16*)alloc((size_t)2 * DI * DM * 2);
  float*    xz      = (float*)   alloc((size_t)MTOT * 2 * DI * 4);
  _Float16* xs_h    = (_Float16*)alloc((size_t)MTOT * DI * 2);
  _Float16* w_xp_h  = (_Float16*)alloc((size_t)NXP * DI * 2);
  float*    dBC     = (float*)   alloc((size_t)MTOT * NXP * 4);
  _Float16* dlt_h   = (_Float16*)alloc((size_t)MTOT * KDT * 2);
  _Float16* w_dt_h  = (_Float16*)alloc((size_t)DI * KDT * 2);
  float*    delta   = (float*)   alloc((size_t)MTOT * DI * 4);
  _Float16* y_h     = (_Float16*)alloc((size_t)MTOT * DI * 2);
  _Float16* w_out_h = (_Float16*)alloc((size_t)DM * DI * 2);
  float*    An      = (float*)   alloc((size_t)DI * DS * 4);
  float*    Aprod   = (float*)   alloc((size_t)BB * DI * NCH * DS * 4);
  float*    Hsum    = (float*)   alloc((size_t)BB * DI * NCH * DS * 4);
  float*    Hinit   = (float*)   alloc((size_t)BB * DI * NCH * DS * 4);

  auto cgrid = [](long long n) { return (unsigned)((n + 255) / 256); };

  // weight conversions (f32 -> f16, padded) + A table
  convert_pad<<<cgrid((long long)2 * DI * DM), 256, 0, stream>>>(in_w, w_in_h, 2 * DI, DM, 2 * DI, DM);
  convert_pad<<<cgrid((long long)NXP * DI),     256, 0, stream>>>(xproj_w, w_xp_h, DTR + 2 * DS, DI, NXP, DI);
  convert_pad<<<cgrid((long long)DI * KDT),     256, 0, stream>>>(dt_w, w_dt_h, DI, DTR, DI, KDT);
  convert_pad<<<cgrid((long long)DM * DI),      256, 0, stream>>>(out_w, w_out_h, DM, DI, DM, DI);
  neg_exp_kernel<<<cgrid(DI * DS), 256, 0, stream>>>(A_log, An);

  // 1) RMSNorm
  rmsnorm_kernel<<<MTOT, 256, 0, stream>>>(x, rms_w, xn_h);

  // 2) in_proj: xz = xn @ in_w^T   (M=4096, N=3072, K=768)
  wmma_gemm<<<dim3((2 * DI) / 128, MTOT / 64), 128, 0, stream>>>(
      xn_h, w_in_h, xz, MTOT, 2 * DI, DM, 0, nullptr);

  // 3) causal depthwise conv + SiLU
  conv_silu_kernel<<<cgrid((long long)MTOT * DI), 256, 0, stream>>>(xz, conv_w, conv_b, xs_h);

  // 4) x_proj: dBC = xs @ xproj_w^T  (N padded 80->128)
  wmma_gemm<<<dim3(NXP / 128, MTOT / 64), 128, 0, stream>>>(
      xs_h, w_xp_h, dBC, MTOT, NXP, DI, 0, nullptr);

  // 5) repack delta_raw (48 -> 64, zero pad)
  pack_delta_kernel<<<cgrid((long long)MTOT * KDT), 256, 0, stream>>>(dBC, dlt_h);

  // 6) dt_proj + bias + softplus: delta = softplus(delta_raw @ dt_w^T + dt_b)
  wmma_gemm<<<dim3(DI / 128, MTOT / 64), 128, 0, stream>>>(
      dlt_h, w_dt_h, delta, MTOT, DI, KDT, 1, dt_b);

  // 7) chunked selective scan + gating -> y_h (f16)
  scan_pass1<<<cgrid((long long)BB * NCH * DI), 256, 0, stream>>>(delta, xs_h, dBC, An, Aprod, Hsum);
  scan_pass2<<<cgrid((long long)BB * DI), 256, 0, stream>>>(Aprod, Hsum, Hinit);
  scan_pass3<<<cgrid((long long)BB * NCH * DI), 256, 0, stream>>>(delta, xs_h, xz, dBC, An, Dvec, Hinit, y_h);

  // 8) out_proj + residual: out = y_gated @ out_w^T + x
  wmma_gemm<<<dim3(DM / 128, MTOT / 64), 128, 0, stream>>>(
      y_h, w_out_h, out, MTOT, DM, DI, 2, x);
}
